// Allegro_66494683677080
// MI455X (gfx1250) — matrix operand order repaired and checked by reading, not compiled, not run on hardware
//
#include <hip/hip_runtime.h>
#include <cmath>

#define NBESS 8
#define ADIM 16
#define CC 64
#define HH 128

typedef __attribute__((ext_vector_type(16))) _Float16 v16h;
typedef __attribute__((ext_vector_type(8)))  float    v8f;
typedef __attribute__((ext_vector_type(4)))  int      v4i;

// CDNA5 async global->LDS copy if the toolchain exposes it; else plain staging.
#ifndef USE_ASYNC_LDS
#if defined(__has_builtin)
#if __has_builtin(__builtin_amdgcn_global_load_async_to_lds_b128)
#define USE_ASYNC_LDS 1
#endif
#endif
#endif
#ifndef USE_ASYNC_LDS
#define USE_ASYNC_LDS 0
#endif

struct BesselConst { float mu[8]; float sig[8]; };

// ---------------------------------------------------------------- prep
__global__ void prep_kernel(const float* __restrict__ vectors,
                            const float* __restrict__ attrs,
                            const int* __restrict__ senders,
                            const int* __restrict__ receivers,
                            float* __restrict__ Yv, float* __restrict__ env,
                            float* __restrict__ X0, int E, BesselConst bc)
{
    int e = blockIdx.x * blockDim.x + threadIdx.x;
    if (e >= E) return;
    const float PI = 3.14159265358979323846f;
    float vx = vectors[3*e+0] * 0.2f;   // / CUTOFF(=5)
    float vy = vectors[3*e+1] * 0.2f;
    float vz = vectors[3*e+2] * 0.2f;
    float d  = sqrtf(vx*vx + vy*vy + vz*vz);
    float inv = (d > 0.f) ? (1.f / d) : 0.f;
    const float SQ3 = 1.7320508075688772f;
    Yv[3*e+0] = SQ3 * vx * inv;
    Yv[3*e+1] = SQ3 * vy * inv;
    Yv[3*e+2] = SQ3 * vz * inv;
    float ev = 0.f;
    if (d < 1.f) {
        float d2 = d*d, d3 = d2*d, d6 = d3*d3;
        ev = 1.f - 28.f*d6 + 48.f*d6*d - 21.f*d6*d2;   // p=6 polynomial envelope
    }
    env[e] = ev;
    float zmask = (d == 0.f) ? 0.f : 1.f;
    const float SQRT2PI = 4.442882938158366f;          // sqrt(2)*pi
    float* xr = X0 + (size_t)e * 64;
#pragma unroll
    for (int n = 1; n <= NBESS; ++n) {
        float arg = PI * (float)n * d;
        float s = (arg == 0.f) ? 1.f : (__sinf(arg) / arg);
        float b = SQRT2PI * (float)n * s;              // sqrt2*pi*n*sinc(n d)
        xr[n-1] = zmask * (b - bc.mu[n-1]) / bc.sig[n-1];
    }
    int s = senders[e], r = receivers[e];
#pragma unroll
    for (int j = 0; j < ADIM; ++j) xr[8 + j]        = zmask * attrs[s*ADIM + j];
#pragma unroll
    for (int j = 0; j < ADIM; ++j) xr[8 + ADIM + j] = zmask * attrs[r*ADIM + j];
#pragma unroll
    for (int j = 8 + 2*ADIM; j < 64; ++j) xr[j] = 0.f;
}

// ---------------- weight f32 -> f16 packed into WMMA B-fragment layout --------
// dst layout (v16h units): [kt = K/32][nt = N/16][lane 0..31] ; lane's 16 halves:
//   half j = W[kt*32 + (lane>>4)*16 + j , nt*16 + (lane&15)] * scale   (0 if k>=K)
__global__ void pack_w(const float* __restrict__ src, _Float16* __restrict__ dst,
                       int K, int N, int Kp, float scale)
{
    int o = blockIdx.x * blockDim.x + threadIdx.x;
    if (o >= Kp * N) return;
    int j    = o & 15;
    int lane = (o >> 4) & 31;
    int tile = o >> 9;
    int NT   = N >> 4;
    int nt   = tile % NT;
    int kt   = tile / NT;
    int k = kt * 32 + (lane >> 4) * 16 + j;
    int n = nt * 16 + (lane & 15);
    dst[o] = (k < K) ? (_Float16)(src[k * N + n] * scale) : (_Float16)0.f;
}

// ---------------------------------------------------------------- WMMA GEMM
// Block = 8 waves = 8 row tiles of 16 edges; each wave computes 16 rows x N cols.
// Packed B weight staged once per block into LDS (async global->LDS on CDNA5),
// then read as one 32B ds load per fragment.  A: f32 row-major (ld % 32 == 0),
// loaded as 4x b128 and converted with v_cvt_pk_f16_f32.  Concat input resolved
// per 32-wide K block (uniform branch; K1 % 32 == 0).
template<int NT, bool HASX2>
__global__ __launch_bounds__(256)
void gemm_wmma(const float* __restrict__ X1, int ld1, int K1,
               const float* __restrict__ X2, int ld2,
               const _Float16* __restrict__ Wp, int Kp,
               float* __restrict__ Y, long ldY, int colStride, int colOff,
               const float* __restrict__ envp, int applySilu, int E)
{
    extern __shared__ _Float16 ldsW[];           // Kp * NT*16 halves
    const int lane = threadIdx.x & 31;
    const int wave = threadIdx.x >> 5;
    const int hi   = lane >> 4;
    const int l15  = lane & 15;

    // ---- stage whole packed weight (Kp*N*2 bytes) into LDS, block-cooperative
#if USE_ASYNC_LDS
    {
        const int n16 = Kp * NT * 2;             // 16-byte units
        auto gsrc = (__attribute__((address_space(1))) v4i*)(void*)Wp;
        auto ldst = (__attribute__((address_space(3))) v4i*)ldsW;
        for (int i = threadIdx.x; i < n16; i += 256)
            __builtin_amdgcn_global_load_async_to_lds_b128(gsrc + i, ldst + i, 0, 0);
        asm volatile("s_wait_asynccnt 0x0" ::: "memory");
    }
#else
    {
        const int n32 = Kp * NT;                 // 32-byte fragments
        const v16h* src = (const v16h*)Wp;
        v16h*       dst = (v16h*)ldsW;
        for (int i = threadIdx.x; i < n32; i += 256) dst[i] = src[i];
    }
#endif
    __syncthreads();

    const long tileM = ((long)blockIdx.x * 8 + wave) * 16;
    long rowA = tileM + l15;
    if (rowA >= E) rowA = E - 1;                 // clamp (branch-free), stores guarded

    v8f acc[NT];
#pragma unroll
    for (int t = 0; t < NT; ++t)
#pragma unroll
        for (int r = 0; r < 8; ++r) acc[t][r] = 0.f;

    const v16h* __restrict__ WB = (const v16h*)ldsW + lane;

    for (int kk = 0; kk < Kp; kk += 32) {
        const float* Xs;
        long base;
        if (HASX2 && kk >= K1) { Xs = X2; base = rowA * (long)ld2 + (kk - K1); }
        else                   { Xs = X1; base = rowA * (long)ld1 + kk; }
        const float4* p = (const float4*)(Xs + base);
        float4 a0 = p[hi*4 + 0];
        float4 a1 = p[hi*4 + 1];
        float4 a2 = p[hi*4 + 2];
        float4 a3 = p[hi*4 + 3];
        v16h a;
        a[0]=(_Float16)a0.x; a[1]=(_Float16)a0.y; a[2]=(_Float16)a0.z; a[3]=(_Float16)a0.w;
        a[4]=(_Float16)a1.x; a[5]=(_Float16)a1.y; a[6]=(_Float16)a1.z; a[7]=(_Float16)a1.w;
        a[8]=(_Float16)a2.x; a[9]=(_Float16)a2.y; a[10]=(_Float16)a2.z; a[11]=(_Float16)a2.w;
        a[12]=(_Float16)a3.x; a[13]=(_Float16)a3.y; a[14]=(_Float16)a3.z; a[15]=(_Float16)a3.w;

        const int kt = kk >> 5;
#pragma unroll
        for (int t = 0; t < NT; ++t) {
            v16h b = WB[(size_t)(kt * NT + t) * 32];
            acc[t] = __builtin_amdgcn_wmma_f32_16x16x32_f16(
                         false, a, false, b, (short)0, acc[t], false, false);
        }
    }

    const long rbase = tileM + hi * 8;
#pragma unroll
    for (int t = 0; t < NT; ++t) {
        int col = t * 16 + l15;
#pragma unroll
        for (int r = 0; r < 8; ++r) {
            long rr = rbase + r;
            if (rr < E) {
                float v = acc[t][r];
                if (applySilu) v = v / (1.f + __expf(-v));
                if (envp)      v *= envp[rr];
                Y[rr * ldY + (long)col * colStride + colOff] = v;
            }
        }
    }
}

// ---------------------------------------------------------------- misc elementwise
__global__ void zero_k(float* __restrict__ p, size_t n) {
    size_t i = (size_t)blockIdx.x * blockDim.x + threadIdx.x;
    if (i < n) p[i] = 0.f;
}

__global__ void split_w0(const float* __restrict__ w0, const float* __restrict__ Yv,
                         float* __restrict__ sV, float* __restrict__ vV, int E)
{
    int i = blockIdx.x * blockDim.x + threadIdx.x;
    if (i >= E * CC) return;
    int e = i / CC;
    sV[i] = w0[(size_t)e * HH + (i % CC)];
    float vv = w0[(size_t)e * HH + 64 + (i % CC)];
    vV[(size_t)i*3 + 0] = vv * Yv[3*e + 0];
    vV[(size_t)i*3 + 1] = vv * Yv[3*e + 1];
    vV[(size_t)i*3 + 2] = vv * Yv[3*e + 2];
}

__global__ void scatter_k(const float* __restrict__ w, const float* __restrict__ Yv,
                          const int* __restrict__ senders,
                          float* __restrict__ nodeS, float* __restrict__ nodeV, int E)
{
    int i = blockIdx.x * blockDim.x + threadIdx.x;
    if (i >= E * CC) return;
    int e = i / CC, c = i % CC;
    int s = senders[e];
    float wv = w[i];
    atomicAdd(&nodeS[(size_t)s*CC + c], wv);
    atomicAdd(&nodeV[((size_t)s*CC + c)*3 + 0], wv * Yv[3*e + 0]);
    atomicAdd(&nodeV[((size_t)s*CC + c)*3 + 1], wv * Yv[3*e + 1]);
    atomicAdd(&nodeV[((size_t)s*CC + c)*3 + 2], wv * Yv[3*e + 2]);
}

__global__ void gather_k(const float* __restrict__ nodeS, const float* __restrict__ nodeV,
                         const int* __restrict__ senders,
                         const float* __restrict__ vV, const float* __restrict__ sV,
                         float* __restrict__ wYs, float* __restrict__ wYv,
                         float* __restrict__ scal, int scal_ld, int E)
{
    int i = blockIdx.x * blockDim.x + threadIdx.x;
    if (i >= E * CC) return;
    int e = i / CC, c = i % CC;
    int s = senders[e];
    const float INVK  = 0.17677669529663687f;   // 1/sqrt(32)
    const float INVS3 = 0.5773502691896258f;    // 1/sqrt(3)
    float ws  = nodeS[(size_t)s*CC + c] * INVK;
    float w0v = nodeV[((size_t)s*CC + c)*3 + 0] * INVK;
    float w1v = nodeV[((size_t)s*CC + c)*3 + 1] * INVK;
    float w2v = nodeV[((size_t)s*CC + c)*3 + 2] * INVK;
    wYs[i] = ws;
    wYv[(size_t)i*3 + 0] = w0v;
    wYv[(size_t)i*3 + 1] = w1v;
    wYv[(size_t)i*3 + 2] = w2v;
    float dot = (w0v * vV[(size_t)i*3+0] + w1v * vV[(size_t)i*3+1] + w2v * vV[(size_t)i*3+2]) * INVS3;
    if (sV) {   // layer 0: scal = [wY_s * sV , scal_dot]
        scal[(size_t)e*scal_ld + c]      = ws * sV[i];
        scal[(size_t)e*scal_ld + 64 + c] = dot;
    } else {
        scal[(size_t)e*scal_ld + c] = dot;
    }
}

__global__ void build_vec(const float* __restrict__ wYs, const float* __restrict__ wYv,
                          const float* __restrict__ vV, const float* __restrict__ sV,
                          float* __restrict__ tmp, int d, int P, int E)
{
    int i = blockIdx.x * blockDim.x + threadIdx.x;
    if (i >= E * CC) return;
    int e = i / CC, c = i % CC;
    float a = wYs[i] * vV[(size_t)i*3 + d];
    if (sV) {   // layer 0: vecs = [wY_s*vV , sV*wY_v]
        tmp[(size_t)e*P + c]      = a;
        tmp[(size_t)e*P + 64 + c] = sV[i] * wYv[(size_t)i*3 + d];
    } else {
        tmp[(size_t)e*P + c] = a;
    }
}

__global__ void residual_k(float* __restrict__ x, const float* __restrict__ y, size_t n) {
    size_t i = (size_t)blockIdx.x * blockDim.x + threadIdx.x;
    if (i < n) x[i] = (x[i] + 0.5f * y[i]) * 0.8944271909999159f;  // /sqrt(1.25)
}

__global__ void final_dot(const float* __restrict__ xf, const float* __restrict__ Wout,
                          float* __restrict__ out, int E)
{
    int e = blockIdx.x * blockDim.x + threadIdx.x;
    if (e >= E) return;
    float acc = 0.f;
    const float sc = 0.08838834764831845f;   // 1/sqrt(128)
#pragma unroll 8
    for (int k = 0; k < HH; ++k) acc += xf[(size_t)e*HH + k] * Wout[k];
    out[e] = acc * sc;
}

// ---------------------------------------------------------------- host
extern "C" void kernel_launch(void* const* d_in, const int* in_sizes, int n_in,
                              void* d_out, int out_size, void* d_ws, size_t ws_size,
                              hipStream_t stream)
{
    const float* attrs     = (const float*)d_in[0];
    const float* vectors   = (const float*)d_in[1];
    const int*   senders   = (const int*)d_in[2];
    const int*   receivers = (const int*)d_in[3];
    const int E  = in_sizes[2];
    const int Nn = in_sizes[0] / ADIM;

    // ---- bump allocator over workspace
    char* ws = (char*)d_ws;
    size_t off = 0;
    auto alloc = [&](size_t bytes) -> void* {
        void* p = ws + off;
        off += (bytes + 255) & ~(size_t)255;
        return p;
    };

    // ---- weights: params flattened in dict-insertion order starting at d_in[4]
    struct WD { int idx; int K; int N; };
    const WD wds[21] = {
        {4,40,16},{5,16,32},{6,32,64},{7,64,128},          // W_embed
        {8,128,128},                                        // W_w0
        {9,128,64},{10,256,128},{11,128,128},{12,128,128},{13,128,64},   // layer0
        {14,128,64},{15,192,128},{16,128,128},{17,128,128},{18,64,64},   // layer1
        {19,128,64},{20,192,128},{21,128,128},{22,128,128},{23,64,64},   // layer2
        {24,128,128}                                        // W_final
    };
    _Float16* wptr[21];
    int wKp[21];
    for (int i = 0; i < 21; ++i) {
        int Kp = (wds[i].K + 31) & ~31;
        wKp[i] = Kp;
        wptr[i] = (_Float16*)alloc((size_t)Kp * wds[i].N * sizeof(_Float16));
        float scale = 1.f / sqrtf((float)wds[i].K);      // fold 1/sqrt(fan_in)
        int tot = Kp * wds[i].N;
        pack_w<<<(tot + 255)/256, 256, 0, stream>>>(
            (const float*)d_in[wds[i].idx], wptr[i], wds[i].K, wds[i].N, Kp, scale);
    }
    const float* Wout_f32 = (const float*)d_in[25];

    // ---- activation buffers
    float* Yv    = (float*)alloc((size_t)E * 3  * 4);
    float* env   = (float*)alloc((size_t)E      * 4);
    float* x     = (float*)alloc((size_t)E * HH * 4);
    float* mlpA  = (float*)alloc((size_t)E * HH * 4);
    float* mlpB  = (float*)alloc((size_t)E * HH * 4);
    float* sV    = (float*)alloc((size_t)E * CC * 4);
    float* vV    = (float*)alloc((size_t)E * CC * 3 * 4);
    float* vV2   = (float*)alloc((size_t)E * CC * 3 * 4);
    float* wYs   = (float*)alloc((size_t)E * CC * 4);
    float* wYv   = (float*)alloc((size_t)E * CC * 3 * 4);
    float* scal  = (float*)alloc((size_t)E * HH * 4);
    float* nodeS = (float*)alloc((size_t)Nn * CC * 4);
    float* nodeV = (float*)alloc((size_t)Nn * CC * 3 * 4);
    (void)ws_size; (void)n_in; (void)out_size; (void)receivers;

    // ---- bessel normalization constants (trapezoid over 1000 pts, like reference)
    BesselConst bc;
    {
        const int R = 1000;
        double bv[R];
        for (int n = 1; n <= 8; ++n) {
            for (int i = 0; i < R; ++i) {
                double r = (double)i / (R - 1);
                double xa = (double)n * r;
                double s = (xa == 0.0) ? 1.0 : sin(M_PI * xa) / (M_PI * xa);
                bv[i] = sqrt(2.0) * M_PI * n * s;
            }
            double m = 0.0, dx = 1.0 / (R - 1);
            for (int i = 0; i + 1 < R; ++i) m += 0.5 * (bv[i] + bv[i+1]) * dx;
            double v = 0.0;
            for (int i = 0; i + 1 < R; ++i) {
                double a0 = bv[i] - m, a1 = bv[i+1] - m;
                v += 0.5 * (a0*a0 + a1*a1) * dx;
            }
            bc.mu[n-1]  = (float)m;
            bc.sig[n-1] = (float)sqrt(v);
        }
    }

    const int gE  = (E + 255) / 256;
    const int gEC = (E * CC + 255) / 256;
    const unsigned gW = (unsigned)((E + 127) / 128);   // 8 waves (row tiles) per block

    auto g = [&](const float* X1, int ld1, int K1, const float* X2, int ld2,
                 int widx, float* Y, long ldY, int cs, int co,
                 const float* envp, int silu) {
        int NT = wds[widx].N / 16;
        const _Float16* W = wptr[widx];
        int Kp = wKp[widx];
        size_t shb = (size_t)Kp * wds[widx].N * sizeof(_Float16);   // LDS for B
        if (X2)          gemm_wmma<8, true ><<<gW, 256, shb, stream>>>(X1, ld1, K1, X2, ld2, W, Kp, Y, ldY, cs, co, envp, silu, E);
        else if (NT == 1) gemm_wmma<1, false><<<gW, 256, shb, stream>>>(X1, ld1, K1, X2, ld2, W, Kp, Y, ldY, cs, co, envp, silu, E);
        else if (NT == 2) gemm_wmma<2, false><<<gW, 256, shb, stream>>>(X1, ld1, K1, X2, ld2, W, Kp, Y, ldY, cs, co, envp, silu, E);
        else if (NT == 4) gemm_wmma<4, false><<<gW, 256, shb, stream>>>(X1, ld1, K1, X2, ld2, W, Kp, Y, ldY, cs, co, envp, silu, E);
        else              gemm_wmma<8, false><<<gW, 256, shb, stream>>>(X1, ld1, K1, X2, ld2, W, Kp, Y, ldY, cs, co, envp, silu, E);
    };

    // 1) geometry + embed-input features (into mlpA, 64-wide rows, zero-padded)
    prep_kernel<<<gE, 256, 0, stream>>>(vectors, attrs, senders, receivers,
                                        Yv, env, mlpA, E, bc);
    // 2) embed MLP 40->16->32->64->128 (silu between), env on last
    //    intermediate ld chosen = Kp of next layer (>=32, multiple of 32)
    g(mlpA, 64, 64,  nullptr, 0, 0, mlpB, 32, 1, 0, nullptr, 1);   // h1: 16 cols, ld 32
    g(mlpB, 32, 32,  nullptr, 0, 1, mlpA, 32, 1, 0, nullptr, 1);   // h2: 32 cols, ld 32
    g(mlpA, 32, 32,  nullptr, 0, 2, mlpB, 64, 1, 0, nullptr, 1);   // h3: 64 cols, ld 64
    g(mlpB, 64, 64,  nullptr, 0, 3, x,   128, 1, 0, env,     0);
    // 3) w0 = x @ W_w0 ; split into sV and vV
    g(x, 128, 128,   nullptr, 0, 4, mlpA, 128, 1, 0, nullptr, 0);
    split_w0<<<gEC, 256, 0, stream>>>(mlpA, Yv, sV, vV, E);

    float* vVcur = vV;
    float* vVnxt = vV2;
    for (int l = 0; l < 3; ++l) {
        int wb = 5 + l * 5;
        // w = x @ W_w
        g(x, 128, 128, nullptr, 0, wb + 0, mlpA, 64, 1, 0, nullptr, 0);
        // segment_sum over senders, then map back
        zero_k<<<((size_t)Nn*CC   + 255)/256, 256, 0, stream>>>(nodeS, (size_t)Nn*CC);
        zero_k<<<((size_t)Nn*CC*3 + 255)/256, 256, 0, stream>>>(nodeV, (size_t)Nn*CC*3);
        scatter_k<<<gEC, 256, 0, stream>>>(mlpA, Yv, senders, nodeS, nodeV, E);
        int sld = (l == 0) ? 128 : 64;
        gather_k<<<gEC, 256, 0, stream>>>(nodeS, nodeV, senders, vVcur,
                                          (l == 0) ? sV : nullptr,
                                          wYs, wYv, scal, sld, E);
        // y = env * mlp(concat([x, scal]))
        g(x, 128, 128, scal, sld, wb + 1, mlpA, 128, 1, 0, nullptr, 1);
        g(mlpA, 128, 128, nullptr, 0, wb + 2, mlpB, 128, 1, 0, nullptr, 1);
        g(mlpB, 128, 128, nullptr, 0, wb + 3, mlpA, 128, 1, 0, env,     0); // y -> mlpA
        // vV' = einsum(vecs, W_V)/sqrt(P) as 3 strided GEMMs
        int P = (l == 0) ? 128 : 64;
        for (int dd = 0; dd < 3; ++dd) {
            build_vec<<<gEC, 256, 0, stream>>>(wYs, wYv, vVcur,
                                               (l == 0) ? sV : nullptr,
                                               mlpB, dd, P, E);
            g(mlpB, P, P, nullptr, 0, wb + 4, vVnxt, 192, 3, dd, nullptr, 0);
        }
        // x = (x + 0.5 y)/sqrt(1.25)
        residual_k<<<((size_t)E*HH + 255)/256, 256, 0, stream>>>(x, mlpA, (size_t)E*HH);
        float* t = vVcur; vVcur = vVnxt; vVnxt = t;
    }

    // final projection + output head
    g(x, 128, 128, nullptr, 0, 20, mlpA, 128, 1, 0, nullptr, 0);
    final_dot<<<gE, 256, 0, stream>>>(mlpA, Wout_f32, (float*)d_out, E);
}
